// PatchScaleLayer_3324304687107
// MI455X (gfx1250) — compile-verified, matrix operand order checked by reading, not run on hardware
//
#include <hip/hip_runtime.h>

// ---------------- model dims ----------------
#define BB 8
#define NN 170
#define LL 336
#define PATCH 16
#define PP 21          // patches per sequence
#define TPAD 32        // padded rows per sequence for WMMA tiles
#define DM 128         // d_model
#define DI 256         // d_inner
#define DS 16          // d_state
#define DTR 8          // dt_rank
#define PRED 96
#define MSEQ (BB*NN)              // 1360
#define ROWS ((size_t)MSEQ*TPAD)  // 43520 padded rows
#define MT  (ROWS/16)             // 2720 row tiles
#define HEADK (PP*DM)             // 2688

typedef __attribute__((ext_vector_type(16))) _Float16 v16h;
typedef __attribute__((ext_vector_type(8)))  float    v8f;

#define WMMA_F16(a, b, c) \
  __builtin_amdgcn_wmma_f32_16x16x32_f16(false, (a), false, (b), (short)0, (c), false, false)

// ---------------- fragment helpers ----------------
// A: 16x32 tile, row-major, leading dim ld (halves). lanes 0-15: row=lane,
// K {0..7,16..23}; lanes 16-31: row=lane-16, K {8..15,24..31}. Two 16B loads.
__device__ __forceinline__ v16h ld_fragA(const _Float16* __restrict__ base, int ld) {
  int lane = threadIdx.x & 31;
  int r  = lane & 15;
  int kb = (lane >> 4) << 3;       // 0 or 8
  const _Float16* p = base + (size_t)r * ld + kb;
  v16h f;
#pragma unroll
  for (int i = 0; i < 8; ++i) { f[i] = p[i]; f[i + 8] = p[i + 16]; }
  return f;
}
// B: packed fragment layout -> one contiguous 32B load per lane.
__device__ __forceinline__ v16h ld_fragB_packed(const _Float16* __restrict__ tileBase) {
  int lane = threadIdx.x & 31;
  return *(const v16h*)(tileBase + lane * 16);
}

// ---------------- kernel 0: W (N,K) fp32 -> fragment-packed f16 B ----------------
// Packed layout: P[((nt*Ktiles + kt)*32 + lane)*16 + i], lane holds col=lane&15,
// kb=(lane>>4)*8; i<8 -> k=kb+i, i>=8 -> k=kb+16+(i-8), within 32-wide K tile.
__global__ void pack_fragB(const float* __restrict__ W, _Float16* __restrict__ P,
                           int N, int K) {
  int idx = blockIdx.x * blockDim.x + threadIdx.x;
  if (idx >= N * K) return;
  int i    = idx & 15;
  int lane = (idx >> 4) & 31;
  int tile = idx >> 9;            // nt*Ktiles + kt
  int Ktiles = K >> 5;
  int nt = tile / Ktiles, kt = tile - nt * Ktiles;
  int c  = lane & 15;
  int kb = (lane >> 4) << 3;
  int kk = (i < 8) ? (kb + i) : (kb + 16 + (i - 8));
  int k = kt * 32 + kk;
  int n = nt * 16 + c;
  P[idx] = (_Float16)W[(size_t)n * K + k];
}

// ---------------- kernel 1: patch-embed + router + LN*gate ----------------
__global__ void encode_kernel(const float* __restrict__ x,
                              const float* __restrict__ nemb,
                              const float* __restrict__ peW,  const float* __restrict__ peb,
                              const float* __restrict__ pos,
                              const float* __restrict__ rW1,  const float* __restrict__ rb1,
                              const float* __restrict__ rW2,  const float* __restrict__ rb2,
                              const float* __restrict__ lng,  const float* __restrict__ lnb,
                              _Float16* __restrict__ XgH, float* __restrict__ Xenc) {
  int m = blockIdx.x;        // 0..1359
  int p = blockIdx.y;        // 0..31
  int d = threadIdx.x;       // 0..127
  size_t rowPad = (size_t)m * TPAD + p;
  if (p >= PP) { XgH[rowPad * DM + d] = (_Float16)0.f; return; }  // uniform per block
  int n = m % NN;

  __shared__ float xp[PATCH];
  __shared__ float energy[9];
  __shared__ float hbuf[32];
  __shared__ float eL[DM];

  if (d < PATCH) xp[d] = x[(size_t)m * LL + p * PATCH + d];
  __syncthreads();

  if (d < 9) {  // 16-point real DFT energies, bins 0..8
    float re = 0.f, im = 0.f;
#pragma unroll
    for (int k = 0; k < PATCH; ++k) {
      float ang = -0.39269908169872414f * (float)(d * k);
      re += xp[k] * __cosf(ang);
      im += xp[k] * __sinf(ang);
    }
    energy[d] = re * re + im * im;
  }

  float e = peb[d] + pos[p * DM + d] + nemb[n * DM + d];
#pragma unroll
  for (int k = 0; k < PATCH; ++k) e += peW[d * PATCH + k] * xp[k];
  eL[d] = e;
  __syncthreads();

  float f0 = energy[0] + energy[1] + energy[2];
  float f1 = energy[3] + energy[4] + energy[5];
  float f2 = energy[6] + energy[7] + energy[8];
  float s = f0 + f1 + f2 + 1e-6f;
  f0 /= s; f1 /= s; f2 /= s;
  if (d < 32) {
    float hv = rW1[d * 3 + 0] * f0 + rW1[d * 3 + 1] * f1 + rW1[d * 3 + 2] * f2 + rb1[d];
    hbuf[d] = hv > 0.f ? hv : 0.f;
  }
  __syncthreads();

  float g = rb2[d];
#pragma unroll
  for (int j = 0; j < 32; ++j) g += rW2[d * 32 + j] * hbuf[j];
  float gate = 1.f / (1.f + __expf(-g));

  float mu = 0.f;
  for (int i = 0; i < DM; ++i) mu += eL[i];
  mu *= (1.f / DM);
  float var = 0.f;
  for (int i = 0; i < DM; ++i) { float t = eL[i] - mu; var += t * t; }
  var *= (1.f / DM);
  float xn = (e - mu) * rsqrtf(var + 1e-5f) * lng[d] + lnb[d];

  XgH[rowPad * DM + d] = (_Float16)(xn * gate);
  Xenc[((size_t)m * PP + p) * DM + d] = e;
}

// ---------------- kernel 2: in_proj GEMM (43520x128x512) ----------------
// grid (MT), block 256 = 8 waves; wave computes a 16x64 strip (4 N-tiles).
__global__ void inproj_gemm(const _Float16* __restrict__ Xg,
                            const _Float16* __restrict__ Pb,     // packed B, Ktiles=4
                            _Float16* __restrict__ XZ) {         // [ROWS][512]
  __shared__ _Float16 Atile[16 * DM];     // 4KB, contiguous copy of A tile
  int tid = threadIdx.x, wave = tid >> 5, lane = tid & 31;
  int mt = blockIdx.x;
  const _Float16* Abase = Xg + (size_t)mt * 16 * DM;   // 2048 contiguous halves
  ((uint4*)Atile)[tid] = ((const uint4*)Abase)[tid];   // 256 x 16B
  __syncthreads();

  v8f acc0 = {}, acc1 = {}, acc2 = {}, acc3 = {};
  int nt0 = wave * 4;                                   // 4 N-tiles per wave
#pragma unroll
  for (int kt = 0; kt < 4; ++kt) {
    v16h a = ld_fragA(Atile + kt * 32, DM);             // ds_load from LDS
    v16h b0 = ld_fragB_packed(Pb + ((size_t)((nt0 + 0) * 4 + kt)) * 512);
    v16h b1 = ld_fragB_packed(Pb + ((size_t)((nt0 + 1) * 4 + kt)) * 512);
    v16h b2 = ld_fragB_packed(Pb + ((size_t)((nt0 + 2) * 4 + kt)) * 512);
    v16h b3 = ld_fragB_packed(Pb + ((size_t)((nt0 + 3) * 4 + kt)) * 512);
    acc0 = WMMA_F16(a, b0, acc0);
    acc1 = WMMA_F16(a, b1, acc1);
    acc2 = WMMA_F16(a, b2, acc2);
    acc3 = WMMA_F16(a, b3, acc3);
  }
  int nlo = lane & 15;
  int mb = (lane >> 4) << 3;
#pragma unroll
  for (int r = 0; r < 8; ++r) {
    size_t row = (size_t)mt * 16 + mb + r;
    XZ[row * 512 + (nt0 + 0) * 16 + nlo] = (_Float16)acc0[r];
    XZ[row * 512 + (nt0 + 1) * 16 + nlo] = (_Float16)acc1[r];
    XZ[row * 512 + (nt0 + 2) * 16 + nlo] = (_Float16)acc2[r];
    XZ[row * 512 + (nt0 + 3) * 16 + nlo] = (_Float16)acc3[r];
  }
}

// ---------------- kernel 3: conv+silu+x_proj+dt+scan+gate ----------------
__global__ void mamba_scan_kernel(const _Float16* __restrict__ XZ,
                                  const float* __restrict__ convW, const float* __restrict__ convB,
                                  const float* __restrict__ xprojW,
                                  const float* __restrict__ dtW,  const float* __restrict__ dtB,
                                  const float* __restrict__ Alog, const float* __restrict__ Dskip,
                                  _Float16* __restrict__ Ys) {    // [m][32][256]
  int m = blockIdx.x;
  int d = threadIdx.x;
  __shared__ float xs[PP * DI];
  __shared__ float proj[PP * 40];
  __shared__ float Bmv[PP * DS];
  __shared__ float Cmv[PP * DS];

  const _Float16* xzm = XZ + (size_t)m * TPAD * 512;

  float w0 = convW[d * 4 + 0], w1 = convW[d * 4 + 1], w2 = convW[d * 4 + 2], w3 = convW[d * 4 + 3];
  float cb = convB[d];
  float xbuf[PP + 3];
  xbuf[0] = xbuf[1] = xbuf[2] = 0.f;
#pragma unroll
  for (int t = 0; t < PP; ++t) xbuf[3 + t] = (float)xzm[t * 512 + d];
#pragma unroll
  for (int t = 0; t < PP; ++t) {
    float c = cb + xbuf[t] * w0 + xbuf[t + 1] * w1 + xbuf[t + 2] * w2 + xbuf[t + 3] * w3;
    xs[t * DI + d] = c * (1.f / (1.f + __expf(-c)));
  }
  __syncthreads();

  for (int o = d; o < PP * 40; o += DI) {
    int t = o / 40, c = o % 40;
    const float* w = xprojW + c * DI;
    const float* xr = xs + t * DI;
    float acc = 0.f;
    for (int k = 0; k < DI; ++k) acc += xr[k] * w[k];
    proj[o] = acc;
  }
  __syncthreads();

  float dtr[PP];
  {
    const float* w = dtW + d * DTR;
    float b0 = dtB[d];
#pragma unroll
    for (int t = 0; t < PP; ++t) {
      float s = b0;
#pragma unroll
      for (int r = 0; r < DTR; ++r) s += proj[t * 40 + r] * w[r];
      dtr[t] = (s > 20.f) ? s : log1pf(__expf(s));
    }
  }
  for (int o = d; o < PP * DS; o += DI) {
    int t = o >> 4, si = o & 15;
    Bmv[o] = proj[t * 40 + DTR + si];
    Cmv[o] = proj[t * 40 + DTR + DS + si];
  }
  __syncthreads();

  float Av[DS], h[DS];
#pragma unroll
  for (int s = 0; s < DS; ++s) { Av[s] = -__expf(Alog[d * DS + s]); h[s] = 0.f; }
  float dsk = Dskip[d];
#pragma unroll 1
  for (int t = 0; t < PP; ++t) {
    float dtt = dtr[t];
    float u = xs[t * DI + d];
    float du = dtt * u;
    float y = 0.f;
#pragma unroll
    for (int s = 0; s < DS; ++s) {
      h[s] = __expf(dtt * Av[s]) * h[s] + du * Bmv[t * DS + s];
      y += h[s] * Cmv[t * DS + s];
    }
    float yv = y + u * dsk;
    float z = (float)xzm[t * 512 + DI + d];
    float zs = z * (1.f / (1.f + __expf(-z)));
    Ys[((size_t)m * TPAD + t) * DI + d] = (_Float16)(yv * zs);
  }
#pragma unroll
  for (int t = PP; t < TPAD; ++t)
    Ys[((size_t)m * TPAD + t) * DI + d] = (_Float16)0.f;
}

// ---------------- kernel 4: out_proj GEMM + residual -> head input ----------------
// grid (MT), block 256 = 8 waves (8 N-tiles cover N=128), Ktiles=8.
__global__ void outproj_gemm(const _Float16* __restrict__ Ysrc,
                             const _Float16* __restrict__ Pb,
                             const float* __restrict__ Xenc,      // [m][21][128]
                             _Float16* __restrict__ Ahead) {      // [1360][2688]
  __shared__ _Float16 Atile[16 * DI];     // 8KB
  int tid = threadIdx.x, wave = tid >> 5, lane = tid & 31;
  int mt = blockIdx.x;
  const _Float16* Abase = Ysrc + (size_t)mt * 16 * DI;  // 4096 contiguous halves
#pragma unroll
  for (int u = 0; u < 2; ++u)
    ((uint4*)Atile)[tid + u * 256] = ((const uint4*)Abase)[tid + u * 256];
  __syncthreads();

  v8f acc = {};
#pragma unroll
  for (int kt = 0; kt < 8; ++kt) {
    v16h a = ld_fragA(Atile + kt * 32, DI);
    v16h b = ld_fragB_packed(Pb + ((size_t)(wave * 8 + kt)) * 512);
    acc = WMMA_F16(a, b, acc);
  }
  int col = wave * 16 + (lane & 15);
  int mb = (lane >> 4) << 3;
#pragma unroll
  for (int r = 0; r < 8; ++r) {
    int grow = mt * 16 + mb + r;
    int m = grow >> 5;
    int t = grow & 31;
    if (t < PP) {
      float v = acc[r] + Xenc[((size_t)m * PP + t) * DM + col];
      Ahead[(size_t)m * HEADK + t * DM + col] = (_Float16)v;
    }
  }
}

// ---------------- kernel 5: head GEMM 1360x2688x96 + bias ----------------
// grid 85, block 192 = 6 waves; A staged in 12KB LDS chunks (7 x 12 K-tiles).
__global__ void head_gemm(const _Float16* __restrict__ Ah,
                          const _Float16* __restrict__ Pb,     // packed B, Ktiles=84
                          const float* __restrict__ hb,
                          float* __restrict__ out) {
  __shared__ _Float16 Atile[16 * 384];    // 12KB: 16 rows x 12 K-tiles
  int tid = threadIdx.x, wave = tid >> 5, lane = tid & 31;
  int mt = blockIdx.x;                    // 0..84
  const _Float16* Abase = Ah + (size_t)mt * 16 * HEADK;
  v8f acc = {};
#pragma unroll 1
  for (int ch = 0; ch < 7; ++ch) {
    __syncthreads();                      // WAR guard on Atile
    for (int u = tid; u < 768; u += 192) {  // 768 x 16B, 4 per thread exactly
      int row = u / 48, cu = u - row * 48;
      *(uint4*)(Atile + row * 384 + cu * 8) =
          *(const uint4*)(Abase + (size_t)row * HEADK + ch * 384 + cu * 8);
    }
    __syncthreads();
#pragma unroll
    for (int kt2 = 0; kt2 < 12; ++kt2) {
      int kt = ch * 12 + kt2;
      v16h a = ld_fragA(Atile + kt2 * 32, 384);
      v16h b = ld_fragB_packed(Pb + ((size_t)(wave * 84 + kt)) * 512);
      acc = WMMA_F16(a, b, acc);
    }
  }
  int col = wave * 16 + (lane & 15);
  int mb = (lane >> 4) << 3;
  float bias = hb[col];
#pragma unroll
  for (int r = 0; r < 8; ++r) {
    int row = mt * 16 + mb + r;           // < 1360 exactly
    out[(size_t)row * PRED + col] = acc[r] + bias;
  }
}

// ---------------- launcher ----------------
extern "C" void kernel_launch(void* const* d_in, const int* in_sizes, int n_in,
                              void* d_out, int out_size, void* d_ws, size_t ws_size,
                              hipStream_t stream) {
  const float* x        = (const float*)d_in[0];
  const float* nemb     = (const float*)d_in[1];
  const float* peW      = (const float*)d_in[2];
  const float* peb      = (const float*)d_in[3];
  const float* pos      = (const float*)d_in[4];
  const float* rW1      = (const float*)d_in[5];
  const float* rb1      = (const float*)d_in[6];
  const float* rW2      = (const float*)d_in[7];
  const float* rb2      = (const float*)d_in[8];
  const float* lng      = (const float*)d_in[9];
  const float* lnb      = (const float*)d_in[10];
  const float* inprojW  = (const float*)d_in[11];
  const float* convW    = (const float*)d_in[12];
  const float* convB    = (const float*)d_in[13];
  const float* xprojW   = (const float*)d_in[14];
  const float* dtW      = (const float*)d_in[15];
  const float* dtB      = (const float*)d_in[16];
  const float* Alog     = (const float*)d_in[17];
  const float* Dskip    = (const float*)d_in[18];
  const float* outprojW = (const float*)d_in[19];
  const float* headW    = (const float*)d_in[20];
  const float* headB    = (const float*)d_in[21];

  char* w = (char*)d_ws;
  size_t off = 0;
  auto take = [&](size_t bytes) -> void* {
    void* p = w + off;
    off += (bytes + 255) & ~(size_t)255;
    return p;
  };

  _Float16* PbIn    = (_Float16*)take((size_t)128 * 512 * 2);
  _Float16* PbOut   = (_Float16*)take((size_t)256 * 128 * 2);
  _Float16* PbHead  = (_Float16*)take((size_t)HEADK * PRED * 2);
  _Float16* XgH     = (_Float16*)take(ROWS * DM * 2);
  float*    Xenc    = (float*)   take((size_t)MSEQ * PP * DM * 4);
  _Float16* XZ      = (_Float16*)take(ROWS * 512 * 2);
  _Float16* Ys      = (_Float16*)take(ROWS * DI * 2);
  _Float16* Ahead   = (_Float16*)take((size_t)MSEQ * HEADK * 2);
  (void)ws_size; (void)in_sizes; (void)n_in; (void)out_size;

  // 0) weights -> fragment-packed f16
  pack_fragB<<<(512 * 128 + 255) / 256, 256, 0, stream>>>(inprojW,  PbIn,  512, 128);
  pack_fragB<<<(128 * 256 + 255) / 256, 256, 0, stream>>>(outprojW, PbOut, 128, 256);
  pack_fragB<<<(PRED * HEADK + 255) / 256, 256, 0, stream>>>(headW, PbHead, PRED, HEADK);

  // 1) encode
  encode_kernel<<<dim3(MSEQ, TPAD), DM, 0, stream>>>(
      x, nemb, peW, peb, pos, rW1, rb1, rW2, rb2, lng, lnb, XgH, Xenc);

  // 2) in_proj WMMA GEMM
  inproj_gemm<<<dim3(MT), 256, 0, stream>>>(XgH, PbIn, XZ);

  // 3) conv + scan
  mamba_scan_kernel<<<MSEQ, DI, 0, stream>>>(XZ, convW, convB, xprojW, dtW, dtB, Alog, Dskip, Ys);

  // 4) out_proj WMMA GEMM + residual
  outproj_gemm<<<dim3(MT), 256, 0, stream>>>(Ys, PbOut, Xenc, Ahead);

  // 5) head WMMA GEMM
  head_gemm<<<85, 192, 0, stream>>>(Ahead, PbHead, headB, (float*)d_out);
}
